// HebbianConv2d_88983132438735
// MI455X (gfx1250) — matrix operand order repaired and checked by reading, not compiled, not run on hardware
//
#include <hip/hip_runtime.h>
#include <hip/hip_bf16.h>

typedef __attribute__((ext_vector_type(16))) _Float16 v16h;
typedef __attribute__((ext_vector_type(8)))  _Float16 v8h;
typedef __attribute__((ext_vector_type(8)))  float    v8f;

#define B_    16
#define CIN   64
#define COUT  128
#define H_    128
#define W_    128
#define HO    126
#define WO    126
#define L_    (HO * WO)        /* 15876 */
#define K_    (CIN * 3 * 3)    /* 576   */
#define NT    32               /* N-tile: spatial positions per block */
#define PK    584              /* padded LDS K stride (halves); 584/8=73 odd -> conflict-free b128 tiling */

// ---------------------------------------------------------------------------
// Kernel 1: per-filter L2 normalization, f32 -> f16, layout [cout][k]
// ---------------------------------------------------------------------------
__global__ __launch_bounds__(256) void hebb_norm_weights(
    const float* __restrict__ w, _Float16* __restrict__ wn)
{
    const int c = blockIdx.x;          // 0..127
    const int t = threadIdx.x;
    __shared__ float red[256];

    float s = 0.f;
    for (int k = t; k < K_; k += 256) {
        float v = w[c * K_ + k];
        s += v * v;
    }
    red[t] = s;
    __syncthreads();
    for (int off = 128; off > 0; off >>= 1) {
        if (t < off) red[t] += red[t + off];
        __syncthreads();
    }
    float nrm = sqrtf(red[0]);
    float inv = (nrm == 0.f) ? 1.f : (1.f / nrm);
    for (int k = t; k < K_; k += 256) {
        wn[c * K_ + k] = (_Float16)(w[c * K_ + k] * inv);
    }
}

// ---------------------------------------------------------------------------
// Kernel 2: implicit-GEMM conv via v_wmma_f32_16x16x32_f16.
// Block = 8 waves, owns NT=32 spatial positions of one image.
// Wave w computes channels [16w,16w+16) x 32 positions: two 16x16 D tiles
// sharing each A (weight) fragment -> 2 WMMA per A-load, 36 WMMA/wave.
// ---------------------------------------------------------------------------
__global__ __launch_bounds__(256) void hebb_conv_wmma(
    const float* __restrict__ x, const _Float16* __restrict__ wn,
    const float* __restrict__ bias, float* __restrict__ y)
{
    __shared__ __attribute__((aligned(64))) _Float16 sp[NT * PK];

    const int bi    = blockIdx.y;
    const int pbase = blockIdx.x * NT;
    const int tid   = threadIdx.x;

    // ---- stage NT x 576 im2col patch tile into LDS as f16 (branch-free) ----
    // thread -> fixed position pos = tid&31 (coalesced wo), k = (tid>>5) + 8*i
    {
        const int pos = tid & 31;
        int p  = pbase + pos;
        int pc = p < L_ ? p : (L_ - 1);            // clamp: tail reads harmless
        int ho = pc / WO, wo = pc - ho * WO;
        const float* xb = x + (size_t)bi * CIN * H_ * W_ + ho * W_ + wo;
        const int kbase = tid >> 5;                 // wave-uniform
        _Float16* spw = &sp[pos * PK];
#pragma unroll 4
        for (int i = 0; i < (NT * K_) / 256; ++i) { // 72 iters
            int k  = kbase + 8 * i;                 // wave-uniform
            int ci = k / 9, r = k - ci * 9;
            int kh = r / 3, kw = r - kh * 3;
            float v = xb[(ci * H_ + kh) * W_ + kw];
            spw[k] = (_Float16)v;
        }
    }
    __syncthreads();

    const int lane  = tid & 31;
    const int wid   = tid >> 5;
    const int lp    = lane & 15;   // M row (A) / N col (B,D)
    const int hi    = lane >> 4;   // half-wave select
    const int mbase = wid * 16;

    const _Float16* wrow = wn + (size_t)(mbase + lp) * K_;
    const _Float16* br0  = &sp[lp * PK];
    const _Float16* br1  = &sp[(16 + lp) * PK];

    // bias folded into accumulator init (D = A*B + C)
    v8f acc0, acc1;
#pragma unroll
    for (int j = 0; j < 8; ++j) {
        float bv = bias[mbase + hi * 8 + j];
        acc0[j] = bv; acc1[j] = bv;
    }

#pragma unroll 3
    for (int kk = 0; kk < K_; kk += 32) {
        // A fragment (16x32 f16): lane row M=lp; halves 0-7 = K kk+8*hi..,
        // halves 8-15 = K kk+16+8*hi..
        v8h a0 = *(const v8h*)(wrow + kk + 8 * hi);
        v8h a1 = *(const v8h*)(wrow + kk + 16 + 8 * hi);
        v16h a = __builtin_shufflevector(a0, a1, 0, 1, 2, 3, 4, 5, 6, 7,
                                                 8, 9, 10, 11, 12, 13, 14, 15);
        // B fragments (32x16 f16): lane col N=lp, same K order as A
        v8h b00 = *(const v8h*)(br0 + kk + 8 * hi);
        v8h b01 = *(const v8h*)(br0 + kk + 16 + 8 * hi);
        v16h b0 = __builtin_shufflevector(b00, b01, 0, 1, 2, 3, 4, 5, 6, 7,
                                                    8, 9, 10, 11, 12, 13, 14, 15);
        v8h b10 = *(const v8h*)(br1 + kk + 8 * hi);
        v8h b11 = *(const v8h*)(br1 + kk + 16 + 8 * hi);
        v16h b1 = __builtin_shufflevector(b10, b11, 0, 1, 2, 3, 4, 5, 6, 7,
                                                    8, 9, 10, 11, 12, 13, 14, 15);
        acc0 = __builtin_amdgcn_wmma_f32_16x16x32_f16(
                   false, a, false, b0, (short)0, acc0, false, false);
        acc1 = __builtin_amdgcn_wmma_f32_16x16x32_f16(
                   false, a, false, b1, (short)0, acc1, false, false);
    }

    // D layout: VGPR j -> M = mbase + j + 8*hi, N = lane%16
    const size_t ybase = (size_t)bi * COUT * L_;
    int p0 = pbase + lp;
    int p1 = p0 + 16;
    if (p0 < L_) {
#pragma unroll
        for (int j = 0; j < 8; ++j)
            y[ybase + (size_t)(mbase + hi * 8 + j) * L_ + p0] = acc0[j];
    }
    if (p1 < L_) {
#pragma unroll
        for (int j = 0; j < 8; ++j)
            y[ybase + (size_t)(mbase + hi * 8 + j) * L_ + p1] = acc1[j];
    }
}

// ---------------------------------------------------------------------------
// Kernel 3: per-(b, c_out) argmax over L (first-index tie-break like argmax)
// ---------------------------------------------------------------------------
__global__ __launch_bounds__(256) void hebb_argmax(
    const float* __restrict__ y, int* __restrict__ winners)
{
    const int bc = blockIdx.x;                 // b*COUT + c
    const float* row = y + (size_t)bc * L_;
    const int t = threadIdx.x;

    float best = -3.402823466e+38f;
    int   bidx = 0;
    for (int i = t; i < L_; i += 256) {
        float v = row[i];
        if (v > best) { best = v; bidx = i; }  // increasing i -> keeps first max
    }
    __shared__ float sv[256];
    __shared__ int   si[256];
    sv[t] = best; si[t] = bidx;
    __syncthreads();
    for (int off = 128; off > 0; off >>= 1) {
        if (t < off) {
            float vo = sv[t + off]; int io = si[t + off];
            if (vo > sv[t] || (vo == sv[t] && io < si[t])) { sv[t] = vo; si[t] = io; }
        }
        __syncthreads();
    }
    if (t == 0) winners[bc] = si[0];
}

// ---------------------------------------------------------------------------
// Kernel 4: delta_w[c][k] = mean over b of winning patch value
// ---------------------------------------------------------------------------
__global__ __launch_bounds__(256) void hebb_delta(
    const float* __restrict__ x, const int* __restrict__ winners,
    float* __restrict__ dw)
{
    const int c = blockIdx.x;                  // 0..127
    __shared__ int wv[B_];
    if (threadIdx.x < B_) wv[threadIdx.x] = winners[threadIdx.x * COUT + c];
    __syncthreads();

    for (int k = threadIdx.x; k < K_; k += 256) {
        int ci = k / 9, r = k - ci * 9;
        int kh = r / 3, kw = r - kh * 3;
        float s = 0.f;
#pragma unroll
        for (int b = 0; b < B_; ++b) {
            int p = wv[b];
            int ho = p / WO, wo = p - ho * WO;
            s += x[(((size_t)b * CIN + ci) * H_ + (ho + kh)) * W_ + (wo + kw)];
        }
        dw[c * K_ + k] = s * (1.f / (float)B_);
    }
}

// ---------------------------------------------------------------------------
extern "C" void kernel_launch(void* const* d_in, const int* in_sizes, int n_in,
                              void* d_out, int out_size, void* d_ws, size_t ws_size,
                              hipStream_t stream)
{
    (void)in_sizes; (void)n_in; (void)out_size; (void)ws_size;

    const float* x    = (const float*)d_in[0];  // (16,64,128,128) f32
    const float* w    = (const float*)d_in[1];  // (128,64,3,3)    f32
    const float* bias = (const float*)d_in[2];  // (128,)          f32

    float* y  = (float*)d_out;                                // 16*128*126*126
    float* dw = y + (size_t)B_ * COUT * L_;                   // 128*64*3*3

    _Float16* wn     = (_Float16*)d_ws;                              // 147456 B
    int*      winner = (int*)((char*)d_ws + (size_t)COUT * K_ * 2);  // 8192 B

    hebb_norm_weights<<<COUT, 256, 0, stream>>>(w, wn);

    dim3 grid((L_ + NT - 1) / NT, B_);   // 497 x 16 blocks
    hebb_conv_wmma<<<grid, 256, 0, stream>>>(x, wn, bias, y);

    hebb_argmax<<<B_ * COUT, 256, 0, stream>>>(y, winner);

    hebb_delta<<<COUT, 256, 0, stream>>>(x, winner, dw);
}
